// UnconditionalMixCDFCouplingBlock_23184233464317
// MI455X (gfx1250) — matrix operand order repaired and checked by reading, not compiled, not run on hardware
//
#include <hip/hip_runtime.h>
#include <hip/hip_bf16.h>

// ---------------------------------------------------------------------------
// Problem constants (from reference): B=8192, D=256, H=1024, K=32, BLOCKS=2
// ---------------------------------------------------------------------------
#define BATCH   8192
#define DFULL   256
#define D2      128
#define HDIM    1024
#define KMIX    32
#define NPAR    98            // 2 + 3*K
#define NOUT    12544         // D2 * NPAR
#define NBLOCKS 2

typedef __bf16 v16bf __attribute__((ext_vector_type(16)));
typedef __bf16 v8bf  __attribute__((ext_vector_type(8)));
typedef float  v8f   __attribute__((ext_vector_type(8)));

static __device__ inline v16bf cat16(v8bf lo, v8bf hi) {
    v16bf r;
#pragma unroll
    for (int i = 0; i < 8; ++i) { r[i] = lo[i]; r[i + 8] = hi[i]; }
    return r;
}
static __device__ inline v8bf ld8(const __bf16* p) {
    return *reinterpret_cast<const v8bf*>(p);
}

// CDNA5 async global->LDS copy (ASYNCcnt-tracked), ISA §10 / §15.18.3:
//   GLOBAL_LOAD_ASYNC_TO_LDS_B128  VDST(lds byte addr), VADDR(64b global), off
static __device__ inline void async_copy_b128(unsigned lds_off, const void* gaddr) {
    asm volatile("global_load_async_to_lds_b128 %0, %1, off"
                 :: "v"(lds_off), "v"((unsigned long long)(uintptr_t)gaddr)
                 : "memory");
}
// Wait variants for double-buffered pipelining (async loads retire in order).
static __device__ inline void async_wait0() {
    asm volatile("s_wait_asynccnt 0x0" ::: "memory");
}
static __device__ inline void async_wait1() {
    asm volatile("s_wait_asynccnt 0x1" ::: "memory");
}
static __device__ inline void async_wait4() {
    asm volatile("s_wait_asynccnt 0x4" ::: "memory");
}
static __device__ inline unsigned lds_off_of(const void* p) {
    return (unsigned)(uintptr_t)p;      // low 32 bits of LDS-aperture flat addr
}

static __device__ inline float wave_max(float v) {
#pragma unroll
    for (int m = 16; m >= 1; m >>= 1) v = fmaxf(v, __shfl_xor(v, m, 32));
    return v;
}
static __device__ inline float wave_sum(float v) {
#pragma unroll
    for (int m = 16; m >= 1; m >>= 1) v += __shfl_xor(v, m, 32);
    return v;
}
static __device__ inline float softplus_f(float a) {     // log(1+exp(a)), stable
    return fmaxf(a, 0.f) + log1pf(expf(-fabsf(a)));
}

// ---------------------------------------------------------------------------
// 0) x1 -> bf16 (WMMA A operand) + copy x1 into output's left half
// ---------------------------------------------------------------------------
__global__ void cvt_x_kernel(const float* __restrict__ x, __bf16* __restrict__ xb,
                             float* __restrict__ y) {
    int i = blockIdx.x * 256 + threadIdx.x;           // over BATCH*D2
    if (i < BATCH * D2) {
        int row = i >> 7, c = i & (D2 - 1);
        float v = x[(size_t)row * DFULL + c];
        xb[i] = (__bf16)v;
        y[(size_t)row * DFULL + c] = v;
    }
}

// ---------------------------------------------------------------------------
// 1) LDS-tiled transpose+convert: src f32 [K][N] row-major -> dst bf16 [N][K]
// ---------------------------------------------------------------------------
__global__ __launch_bounds__(256) void transpose_cvt_kernel(
        const float* __restrict__ src, __bf16* __restrict__ dst, int K, int N) {
    __shared__ float tile[32][33];
    int nb = blockIdx.x * 32, kb = blockIdx.y * 32;
    int tx = threadIdx.x & 31, ty = threadIdx.x >> 5;       // 32 x 8
#pragma unroll
    for (int i = 0; i < 32; i += 8) {
        int k = kb + ty + i, n = nb + tx;
        tile[ty + i][tx] = (k < K && n < N) ? src[(size_t)k * N + n] : 0.f;
    }
    __syncthreads();
#pragma unroll
    for (int i = 0; i < 32; i += 8) {
        int n = nb + ty + i, k = kb + tx;
        if (n < N && k < K) dst[(size_t)n * K + k] = (__bf16)tile[tx][ty + i];
    }
}

// ---------------------------------------------------------------------------
// 2) WMMA GEMM:  out = act(A[M,Kd] @ Bt[N,Kd]^T + bias (+ resid))
//    256 threads = 8 waves (2 wave-rows x 4 wave-cols).
//    WG tile M=64 x N=256; wave tile 32x64 (2x4 accumulators, 8 WMMA/k-step).
//    A chunks (64 x 128 bf16 = 16KB) double-buffered in LDS via async copies:
//    chunk i+1 stays in flight (s_wait_asynccnt 4) under chunk i's WMMA loop.
// ---------------------------------------------------------------------------
#define GEMM_KC 128
__global__ __launch_bounds__(256) void gemm_bf16_kernel(
        const __bf16* __restrict__ A, const __bf16* __restrict__ Bt,
        const float* __restrict__ bias, const float* __restrict__ resid,
        float* __restrict__ outF, __bf16* __restrict__ outB,
        int N, int Kd, int relu) {
    __shared__ __bf16 aT[2][64 * GEMM_KC];                 // 2 x 16 KB
    int tid  = threadIdx.x;
    int w    = tid >> 5, lane = tid & 31;
    int wr   = w >> 2,  wc   = w & 3;
    int mbase = blockIdx.x * 64;
    int m0    = mbase + wr * 32;
    int n0    = blockIdx.y * 256 + wc * 64;
    int half  = lane >> 4, l15 = lane & 15;

    v8f acc[2][4];
#pragma unroll
    for (int i = 0; i < 2; ++i)
#pragma unroll
        for (int nt = 0; nt < 4; ++nt) acc[i][nt] = 0.f;

    const __bf16* brow[4];
#pragma unroll
    for (int nt = 0; nt < 4; ++nt)
        brow[nt] = Bt + (size_t)(n0 + nt * 16 + l15) * Kd + half * 16;

    // staging map: thread -> (row, 64B quarter); 4 async b128 per thread/chunk
    int srow = tid >> 2, sq = tid & 3;
    const __bf16* gsrc = A + (size_t)(mbase + srow) * Kd + sq * 32;
    unsigned lbase[2] = { lds_off_of(&aT[0][srow * GEMM_KC + sq * 32]),
                          lds_off_of(&aT[1][srow * GEMM_KC + sq * 32]) };

    int nchunks = Kd / GEMM_KC;
    // prologue: stage chunk 0 into buffer 0
    async_copy_b128(lbase[0],      gsrc);
    async_copy_b128(lbase[0] + 16, gsrc + 8);
    async_copy_b128(lbase[0] + 32, gsrc + 16);
    async_copy_b128(lbase[0] + 48, gsrc + 24);

    for (int ci = 0; ci < nchunks; ++ci) {
        int kc = ci * GEMM_KC;
        if (ci + 1 < nchunks) {                            // stage next chunk
            const __bf16* g = gsrc + kc + GEMM_KC;
            unsigned lo = lbase[(ci + 1) & 1];
            async_copy_b128(lo,      g);
            async_copy_b128(lo + 16, g + 8);
            async_copy_b128(lo + 32, g + 16);
            async_copy_b128(lo + 48, g + 24);
            async_wait4();                                 // chunk ci done, ci+1 in flight
        } else {
            async_wait0();
        }
        __syncthreads();
        const __bf16* aTr0 = &aT[ci & 1][(wr * 32 + l15) * GEMM_KC];
        const __bf16* aTr1 = aTr0 + 16 * GEMM_KC;
#pragma unroll
        for (int k = 0; k < GEMM_KC; k += 32) {
            v16bf a0 = cat16(ld8(aTr0 + k + half * 16), ld8(aTr0 + k + half * 16 + 8));
            v16bf a1 = cat16(ld8(aTr1 + k + half * 16), ld8(aTr1 + k + half * 16 + 8));
#pragma unroll
            for (int nt = 0; nt < 4; ++nt) {
                v16bf b = cat16(ld8(brow[nt] + kc + k), ld8(brow[nt] + kc + k + 8));
                acc[0][nt] = __builtin_amdgcn_wmma_f32_16x16x32_bf16(
                                 false, a0, false, b, (short)0, acc[0][nt], false, false);
                acc[1][nt] = __builtin_amdgcn_wmma_f32_16x16x32_bf16(
                                 false, a1, false, b, (short)0, acc[1][nt], false, false);
            }
        }
        __syncthreads();
    }

#pragma unroll
    for (int i = 0; i < 2; ++i) {
#pragma unroll
        for (int nt = 0; nt < 4; ++nt) {
#pragma unroll
            for (int j = 0; j < 8; ++j) {
                int row = m0 + i * 16 + j + 8 * half;   // C/D layout
                int col = n0 + nt * 16 + l15;
                size_t idx = (size_t)row * N + col;
                float v = acc[i][nt][j] + bias[col];
                if (resid) v += resid[idx];
                if (relu)  v = fmaxf(v, 0.f);
                if (outF)  outF[idx] = v;
                if (outB)  outB[idx] = (__bf16)v;
            }
        }
    }
}

// ---------------------------------------------------------------------------
// 3) Fused output projection + mixture-CDF coupling epilogue.
//    WG: 16 rows x 8 d2-groups (wave = one d2 group). The shared 16-row A
//    slab is double-buffered in LDS (async, 1 b128/thread/chunk) and read by
//    all 8 waves. Each wave: 7 WMMA accumulators -> 16x98 param slab -> LDS
//    -> epilogue with lane == mixture component (K=32 == wave32).
// ---------------------------------------------------------------------------
#define FUSE_KC 128
__global__ __launch_bounds__(256) void fused_out_kernel(
        const __bf16* __restrict__ hb, const __bf16* __restrict__ Wt,
        const float* __restrict__ b_out, const float* __restrict__ x,
        float* __restrict__ y, float* __restrict__ ldbuf) {
    __shared__ __bf16 aT[2][16 * FUSE_KC];                 // 2 x 4 KB
    __shared__ float  pls[8][16][104];                     // 52 KB
    int tid  = threadIdx.x;
    int w    = tid >> 5, lane = tid & 31;
    int half = lane >> 4, l15 = lane & 15;
    int rowbase = blockIdx.x * 16;
    int g = blockIdx.y * 8 + w;                            // d2 index, 0..127
    const int Kd = HDIM;

    v8f acc[7];
#pragma unroll
    for (int i = 0; i < 7; ++i) acc[i] = 0.f;

    int ncol = g * NPAR + l15;
    int srow = tid >> 4, seg = tid & 15;                   // 1 async b128 per thread
    const __bf16* gsrc = hb + (size_t)(rowbase + srow) * Kd + seg * 8;
    unsigned lbase[2] = { lds_off_of(&aT[0][srow * FUSE_KC + seg * 8]),
                          lds_off_of(&aT[1][srow * FUSE_KC + seg * 8]) };

    const int nchunks = Kd / FUSE_KC;
    async_copy_b128(lbase[0], gsrc);                       // prologue: chunk 0

    for (int ci = 0; ci < nchunks; ++ci) {
        int kc = ci * FUSE_KC;
        if (ci + 1 < nchunks) {
            async_copy_b128(lbase[(ci + 1) & 1], gsrc + kc + FUSE_KC);
            async_wait1();                                 // chunk ci done, ci+1 in flight
        } else {
            async_wait0();
        }
        __syncthreads();
        const __bf16* aTr = &aT[ci & 1][l15 * FUSE_KC];
#pragma unroll
        for (int k = 0; k < FUSE_KC; k += 32) {
            v16bf a = cat16(ld8(aTr + k + half * 16), ld8(aTr + k + half * 16 + 8));
#pragma unroll
            for (int nt = 0; nt < 7; ++nt) {
                int n = ncol + nt * 16;
                n = (n < NOUT) ? n : (NOUT - 1);           // pad cols, discarded later
                const __bf16* br = Wt + (size_t)n * Kd + kc + k + half * 16;
                v16bf b = cat16(ld8(br), ld8(br + 8));
                acc[nt] = __builtin_amdgcn_wmma_f32_16x16x32_bf16(
                              false, a, false, b, (short)0, acc[nt], false, false);
            }
        }
        __syncthreads();
    }

    // Park the 16x98 slab (+bias) in LDS, one sub-array per wave.
#pragma unroll
    for (int nt = 0; nt < 7; ++nt) {
        int c = nt * 16 + l15;
#pragma unroll
        for (int j = 0; j < 8; ++j) {
            int r = j + 8 * half;
            if (c < NPAR) pls[w][r][c] = acc[nt][j] + b_out[g * NPAR + c];
        }
    }
    __syncthreads();

    // Epilogue: lane = mixture component.
    int kk = lane;
    for (int r = 0; r < 16; ++r) {
        float s     = tanhf(pls[w][r][0]);
        float t     = pls[w][r][1];
        float logit = pls[w][r][2 + kk];
        float mean  = pls[w][r][2 + KMIX + kk];
        float lsc   = fmaxf(pls[w][r][2 + 2 * KMIX + kk], -7.f);
        float x2    = x[(size_t)(rowbase + r) * DFULL + D2 + g];

        float mx  = wave_max(logit);
        float lse = mx + logf(wave_sum(expf(logit - mx)));
        float lsm = logit - lse;                           // log_softmax

        float z  = (x2 - mean) * expf(-lsc);
        float sp = softplus_f(-z);                         // softplus(-z)

        float tc = lsm - sp;                               // + log_sigmoid(z)
        float m2 = wave_max(tc);
        float logcdf = m2 + logf(wave_sum(expf(tc - m2)));

        float tp = lsm + (-z - lsc - 2.f * sp);            // + log_pdf component
        float m3 = wave_max(tp);
        float logpdf = m3 + logf(wave_sum(expf(tp - m3)));

        if (kk == 0) {
            float u      = expf(logcdf);
            float log1mu = log1pf(-u);
            float y2     = (logcdf - log1mu) * expf(s) + t;   // logit(u)*exp(s)+t
            y[(size_t)(rowbase + r) * DFULL + D2 + g] = y2;
            ldbuf[(size_t)(rowbase + r) * D2 + g] = logpdf + s - logcdf - log1mu;
        }
    }
}

// ---------------------------------------------------------------------------
// 4) Deterministic logdet reduction: sum 128 contributions per row.
// ---------------------------------------------------------------------------
__global__ void reduce_logdet_kernel(const float* __restrict__ ldbuf,
                                     float* __restrict__ ld) {
    int row = blockIdx.x * 256 + threadIdx.x;
    if (row < BATCH) {
        float s = 0.f;
        const float* p = ldbuf + (size_t)row * D2;
#pragma unroll 8
        for (int g = 0; g < D2; ++g) s += p[g];
        ld[row] = s;
    }
}

// ---------------------------------------------------------------------------
// Host launcher
// ---------------------------------------------------------------------------
extern "C" void kernel_launch(void* const* d_in, const int* in_sizes, int n_in,
                              void* d_out, int out_size, void* d_ws, size_t ws_size,
                              hipStream_t stream) {
    const float* x      = (const float*)d_in[0];
    const float* W_in   = (const float*)d_in[1];
    const float* b_in   = (const float*)d_in[2];
    const float* W_blk1 = (const float*)d_in[3];
    const float* b_blk1 = (const float*)d_in[4];
    const float* W_blk2 = (const float*)d_in[5];
    const float* b_blk2 = (const float*)d_in[6];
    const float* W_out  = (const float*)d_in[7];
    const float* b_out  = (const float*)d_in[8];
    float* yout = (float*)d_out;                       // [B,256] then logdet [B]
    float* ldet = yout + (size_t)BATCH * DFULL;

    auto align256 = [](size_t v) { return (v + 255) & ~(size_t)255; };
    char*  ws  = (char*)d_ws;
    size_t off = 0;
    __bf16* xb    = (__bf16*)(ws + off); off = align256(off + (size_t)BATCH * D2 * 2);
    __bf16* wtin  = (__bf16*)(ws + off); off = align256(off + (size_t)HDIM * D2 * 2);
    __bf16* wtb1  = (__bf16*)(ws + off); off = align256(off + (size_t)NBLOCKS * HDIM * HDIM * 2);
    __bf16* wtb2  = (__bf16*)(ws + off); off = align256(off + (size_t)NBLOCKS * HDIM * HDIM * 2);
    __bf16* wtout = (__bf16*)(ws + off); off = align256(off + (size_t)NOUT * HDIM * 2);
    float*  hbuf  = (float*)(ws + off);  off = align256(off + (size_t)BATCH * HDIM * 4);
    __bf16* hb    = (__bf16*)(ws + off); off = align256(off + (size_t)BATCH * HDIM * 2);
    __bf16* rb    = (__bf16*)(ws + off); off = align256(off + (size_t)BATCH * HDIM * 2);
    float*  ldbuf = (float*)(ws + off);  off = align256(off + (size_t)BATCH * D2 * 4);
    (void)ws_size; (void)in_sizes; (void)n_in; (void)out_size;

    // 0) x1 -> bf16 + copy left half of output
    cvt_x_kernel<<<(BATCH * D2 + 255) / 256, 256, 0, stream>>>(x, xb, yout);

    // 1) Transpose-convert all weights to bf16 [N][K]
    transpose_cvt_kernel<<<dim3(HDIM / 32, (D2 + 31) / 32), 256, 0, stream>>>(
        W_in, wtin, D2, HDIM);
    for (int i = 0; i < NBLOCKS; ++i) {
        transpose_cvt_kernel<<<dim3(HDIM / 32, HDIM / 32), 256, 0, stream>>>(
            W_blk1 + (size_t)i * HDIM * HDIM, wtb1 + (size_t)i * HDIM * HDIM, HDIM, HDIM);
        transpose_cvt_kernel<<<dim3(HDIM / 32, HDIM / 32), 256, 0, stream>>>(
            W_blk2 + (size_t)i * HDIM * HDIM, wtb2 + (size_t)i * HDIM * HDIM, HDIM, HDIM);
    }
    transpose_cvt_kernel<<<dim3((NOUT + 31) / 32, HDIM / 32), 256, 0, stream>>>(
        W_out, wtout, HDIM, NOUT);

    // 2) h = relu(x1 @ W_in + b_in)
    gemm_bf16_kernel<<<dim3(BATCH / 64, HDIM / 256), 256, 0, stream>>>(
        xb, wtin, b_in, nullptr, hbuf, hb, HDIM, D2, 1);

    // 3) residual blocks
    for (int i = 0; i < NBLOCKS; ++i) {
        gemm_bf16_kernel<<<dim3(BATCH / 64, HDIM / 256), 256, 0, stream>>>(
            hb, wtb1 + (size_t)i * HDIM * HDIM, b_blk1 + i * HDIM,
            nullptr, nullptr, rb, HDIM, HDIM, 1);
        gemm_bf16_kernel<<<dim3(BATCH / 64, HDIM / 256), 256, 0, stream>>>(
            rb, wtb2 + (size_t)i * HDIM * HDIM, b_blk2 + i * HDIM,
            hbuf, hbuf, hb, HDIM, HDIM, 0);
    }

    // 4) fused output GEMM + mixture coupling epilogue
    fused_out_kernel<<<dim3(BATCH / 16, D2 / 8), 256, 0, stream>>>(
        hb, wtout, b_out, x, yout, ldbuf);

    // 5) deterministic logdet reduction
    reduce_logdet_kernel<<<(BATCH + 255) / 256, 256, 0, stream>>>(ldbuf, ldet);
}
